// Attention_33062658244770
// MI455X (gfx1250) — matrix operand order repaired and checked by reading, not compiled, not run on hardware
//
#include <hip/hip_runtime.h>
#include <hip/hip_bf16.h>

#define B_   2
#define S_   2048
#define D_   2048
#define H_   16
#define KV_  4
#define HD_  128
#define REP_ 4

typedef __attribute__((ext_vector_type(16))) __bf16       v16bf;
typedef __attribute__((ext_vector_type(8)))  float        v8f;
typedef __attribute__((ext_vector_type(4)))  unsigned int u32x4;

union Frag16 {
  u32x4          q[2];
  v16bf          v;
  unsigned short s[16];
};

__device__ __forceinline__ unsigned short f2bf(float f) {
  union { float f; unsigned u; } c; c.f = f;
  unsigned u = c.u;
  u += 0x7fffu + ((u >> 16) & 1u);   // round-to-nearest-even
  return (unsigned short)(u >> 16);
}

// Async copy 16B global -> LDS (GLOBAL_LOAD_ASYNC_TO_LDS_B128, ASYNCcnt).
__device__ __forceinline__ void async_copy16(unsigned lds_off,
                                             const unsigned short* g) {
  asm volatile("global_load_async_to_lds_b128 %0, %1, off"
               :: "v"(lds_off), "v"(g) : "memory");
}
__device__ __forceinline__ void async_wait_all() {
  asm volatile("s_wait_asynccnt 0x0" ::: "memory");
}

#define WMMA_BF16(A, B, C)                                                   \
  __builtin_amdgcn_wmma_f32_16x16x32_bf16(false, (A), false, (B), (short)0,  \
                                          (C), false, false)

// ---------------------------------------------------------------- fp32 -> bf16
__global__ void cvt_bf16_kernel(const float* __restrict__ in,
                                unsigned short* __restrict__ out, int n) {
  int i = blockIdx.x * blockDim.x + threadIdx.x;
  int stride = gridDim.x * blockDim.x;
  for (; i < n; i += stride) out[i] = f2bf(in[i]);
}

// ------------------------------------------------- fused QKV GEMM + RoPE
// Block: 256 thr = 8 waves (2 M x 4 N), block tile 64(M) x 128(N), wave 32x32.
// A (x rows) staged in LDS via async copies, double buffered, Kc = 128.
__global__ __launch_bounds__(256)
void qkv_rope_kernel(const unsigned short* __restrict__ xb,
                     const unsigned short* __restrict__ wqb,
                     const unsigned short* __restrict__ wkb,
                     const unsigned short* __restrict__ wvb,
                     const float* __restrict__ fc,
                     const float* __restrict__ fs,
                     unsigned short* __restrict__ qo,    // [B,H,S,HD]
                     unsigned short* __restrict__ ko,    // [B,KV,S,HD]
                     unsigned short* __restrict__ vto) { // [B,KV,HD,S]
  __shared__ __align__(16) unsigned short ldsA[2][64][128];

  const int tid  = threadIdx.x;
  const int lane = tid & 31;
  const int wid  = tid >> 5;
  const int l16  = lane & 15;
  const int h    = lane >> 4;
  const int wm   = wid >> 2;   // 0..1
  const int wn   = wid & 3;    // 0..3

  const int m_blk = blockIdx.x * 64;
  const int n0w   = blockIdx.y * 128 + wn * 32;   // region boundaries are /128

  const unsigned short* wbase;
  int nloc;
  if (n0w < D_)            { wbase = wqb; nloc = n0w; }
  else if (n0w < D_ + 512) { wbase = wkb; nloc = n0w - D_; }
  else                     { wbase = wvb; nloc = n0w - (D_ + 512); }

  const unsigned short* brow0 = wbase + (size_t)(nloc + l16) * D_;
  const unsigned short* brow1 = wbase + (size_t)(nloc + 16 + l16) * D_;

  auto stage = [&](int buf, int k0) {   // 64 rows x 128 bf16 = 1024 x 16B
#pragma unroll
    for (int c = 0; c < 4; ++c) {
      const int chunk = tid * 4 + c;        // 0..1023
      const int row   = chunk >> 3;         // 8 chunks per row
      const int off   = (chunk & 7) * 8;    // element offset
      const unsigned short* g = xb + (size_t)(m_blk + row) * D_ + k0 + off;
      const unsigned laddr = (unsigned)(size_t)&ldsA[buf][row][off];
      async_copy16(laddr, g);
    }
  };

  v8f acc[2][2] = {};

  stage(0, 0);
  async_wait_all();
  __syncthreads();

  const int NS = D_ / 128;                  // 16 k-slices
  for (int s = 0; s < NS; ++s) {
    const int k0  = s * 128;
    const int cur = s & 1;
    if (s + 1 < NS) stage(cur ^ 1, k0 + 128);

#pragma unroll
    for (int kk = 0; kk < 128; kk += 32) {
      Frag16 a0, a1, b0, b1;
      const unsigned short* l0 = &ldsA[cur][wm * 32 + l16][kk];
      const unsigned short* l1 = &ldsA[cur][wm * 32 + 16 + l16][kk];
      a0.q[0] = *reinterpret_cast<const u32x4*>(l0 + 16 * h);
      a0.q[1] = *reinterpret_cast<const u32x4*>(l0 + 16 * h + 8);
      a1.q[0] = *reinterpret_cast<const u32x4*>(l1 + 16 * h);
      a1.q[1] = *reinterpret_cast<const u32x4*>(l1 + 16 * h + 8);
      b0.q[0] = *reinterpret_cast<const u32x4*>(brow0 + k0 + kk + 16 * h);
      b0.q[1] = *reinterpret_cast<const u32x4*>(brow0 + k0 + kk + 16 * h + 8);
      b1.q[0] = *reinterpret_cast<const u32x4*>(brow1 + k0 + kk + 16 * h);
      b1.q[1] = *reinterpret_cast<const u32x4*>(brow1 + k0 + kk + 16 * h + 8);
      acc[0][0] = WMMA_BF16(a0.v, b0.v, acc[0][0]);
      acc[0][1] = WMMA_BF16(a0.v, b1.v, acc[0][1]);
      acc[1][0] = WMMA_BF16(a1.v, b0.v, acc[1][0]);
      acc[1][1] = WMMA_BF16(a1.v, b1.v, acc[1][1]);
    }
    async_wait_all();
    __syncthreads();
  }

  // Epilogue: RoPE for Q/K, transpose-store for V.
#pragma unroll
  for (int im = 0; im < 2; ++im) {
#pragma unroll
    for (int in_ = 0; in_ < 2; ++in_) {
      const int n = n0w + in_ * 16 + l16;
#pragma unroll
      for (int r = 0; r < 8; ++r) {
        const int m    = m_blk + wm * 32 + im * 16 + r + 8 * h;
        const int pos  = m & (S_ - 1);
        const int bidx = m >> 11;
        float val      = acc[im][in_][r];
        float partner  = __shfl_xor(val, 1, 32);   // even<->odd channel
        if (n < D_ + 512) {                        // Q or K: RoPE
          const int nl = (n < D_) ? n : (n - D_);
          const int dd = nl & (HD_ - 1);
          const int p  = dd >> 1;
          const float c = fc[pos * (HD_ / 2) + p];
          const float sn = fs[pos * (HD_ / 2) + p];
          const float rv = ((dd & 1) == 0) ? (val * c - partner * sn)
                                           : (partner * sn + val * c);
          if (n < D_) {
            const int head = n >> 7;
            qo[((size_t)(bidx * H_ + head) * S_ + pos) * HD_ + dd] = f2bf(rv);
          } else {
            const int gk = nl >> 7;
            ko[((size_t)(bidx * KV_ + gk) * S_ + pos) * HD_ + dd] = f2bf(rv);
          }
        } else {                                   // V: store transposed
          const int nv = n - (D_ + 512);
          const int gv = nv >> 7;
          const int dd = nv & (HD_ - 1);
          vto[((size_t)(bidx * KV_ + gv) * HD_ + dd) * S_ + pos] = f2bf(val);
        }
      }
    }
  }
}

// --------------------------------------------- flash attention (1 wave / 16 q)
// S^T = K*Q^T so the softmax'd P^T C-fragment feeds PV's B-fragment directly.
__global__ __launch_bounds__(128)
void attn_fa_kernel(const unsigned short* __restrict__ qb,   // [B,H,S,HD]
                    const unsigned short* __restrict__ kbuf, // [B,KV,S,HD]
                    const unsigned short* __restrict__ vtb,  // [B,KV,HD,S]
                    unsigned short* __restrict__ attnb) {    // [B,S,H*HD]
  const int lane = threadIdx.x & 31;
  const int wid  = threadIdx.x >> 5;
  const int l16  = lane & 15;
  const int h    = lane >> 4;

  const int qt = blockIdx.x * 4 + wid;
  const int bh = blockIdx.y;
  const int b  = bh / H_;
  const int hh = bh - b * H_;
  const int g  = hh / REP_;

  const int q0 = qt * 16;
  const int qg = q0 + l16;

  const unsigned short* qrow = qb + ((size_t)(b * H_ + hh) * S_ + qg) * HD_;
  Frag16 qf[4];
#pragma unroll
  for (int t = 0; t < 4; ++t) {
    qf[t].q[0] = *reinterpret_cast<const u32x4*>(qrow + t * 32 + 8 * h);
    qf[t].q[1] = *reinterpret_cast<const u32x4*>(qrow + t * 32 + 16 + 8 * h);
  }

  const unsigned short* kb0 = kbuf + (size_t)(b * KV_ + g) * S_ * HD_;
  const unsigned short* vb0 = vtb  + (size_t)(b * KV_ + g) * HD_ * S_;

  float o[8][8];
#pragma unroll
  for (int i = 0; i < 8; ++i)
#pragma unroll
    for (int j = 0; j < 8; ++j) o[i][j] = 0.0f;

  float m_run = -3.0e38f;
  float l_run = 0.0f;
  const float sc = 0.088388347648318447f; // 1/sqrt(128)

  const int nblk = (q0 + 15) / 32 + 1;
  for (int blk = 0; blk < nblk; ++blk) {
    const int kb = blk * 32;

    v8f s0 = {}, s1 = {};
#pragma unroll
    for (int t = 0; t < 4; ++t) {
      Frag16 a0, a1;
      const unsigned short* kr0 = kb0 + (size_t)(kb + l16) * HD_ + t * 32;
      const unsigned short* kr1 = kb0 + (size_t)(kb + 16 + l16) * HD_ + t * 32;
      a0.q[0] = *reinterpret_cast<const u32x4*>(kr0 + 8 * h);
      a0.q[1] = *reinterpret_cast<const u32x4*>(kr0 + 16 + 8 * h);
      a1.q[0] = *reinterpret_cast<const u32x4*>(kr1 + 8 * h);
      a1.q[1] = *reinterpret_cast<const u32x4*>(kr1 + 16 + 8 * h);
      s0 = WMMA_BF16(a0.v, qf[t].v, s0);
      s1 = WMMA_BF16(a1.v, qf[t].v, s1);
    }

    float f0[8], f1[8];
    float mx = -3.0e38f;
#pragma unroll
    for (int r = 0; r < 8; ++r) {
      const int k0i = kb + 8 * h + r;
      const int k1i = k0i + 16;
      f0[r] = (k0i <= qg) ? s0[r] * sc : -3.0e38f;
      f1[r] = (k1i <= qg) ? s1[r] * sc : -3.0e38f;
      mx = fmaxf(mx, fmaxf(f0[r], f1[r]));
    }
    mx = fmaxf(mx, __shfl_xor(mx, 16, 32));
    const float mnew  = fmaxf(m_run, mx);
    const float alpha = __expf(m_run - mnew);

    float p0[8], p1[8];
    float lsum = 0.0f;
#pragma unroll
    for (int r = 0; r < 8; ++r) {
      p0[r] = __expf(f0[r] - mnew);
      p1[r] = __expf(f1[r] - mnew);
      lsum += p0[r] + p1[r];
    }
    lsum += __shfl_xor(lsum, 16, 32);
    l_run = l_run * alpha + lsum;
    m_run = mnew;

#pragma unroll
    for (int i = 0; i < 8; ++i)
#pragma unroll
      for (int j = 0; j < 8; ++j) o[i][j] *= alpha;

    float pp0[8], pp1[8];
#pragma unroll
    for (int r = 0; r < 8; ++r) {
      pp0[r] = __shfl_xor(p0[r], 16, 32);
      pp1[r] = __shfl_xor(p1[r], 16, 32);
    }
    Frag16 pf;
#pragma unroll
    for (int i = 0; i < 8; ++i) {
      pf.s[i]     = f2bf(h == 0 ? p0[i]  : pp1[i]);
      pf.s[i + 8] = f2bf(h == 0 ? pp0[i] : p1[i]);
    }

#pragma unroll
    for (int dt = 0; dt < 8; ++dt) {
      Frag16 va;
      const unsigned short* vr = vb0 + (size_t)(dt * 16 + l16) * S_ + kb;
      va.q[0] = *reinterpret_cast<const u32x4*>(vr + 16 * h);
      va.q[1] = *reinterpret_cast<const u32x4*>(vr + 16 * h + 8);
      v8f cacc;
#pragma unroll
      for (int e = 0; e < 8; ++e) cacc[e] = o[dt][e];
      cacc = WMMA_BF16(va.v, pf.v, cacc);
#pragma unroll
      for (int e = 0; e < 8; ++e) o[dt][e] = cacc[e];
    }
  }

  const float inv = (l_run > 0.0f) ? (1.0f / l_run) : 0.0f;
  unsigned short* orow = attnb + (size_t)(b * S_ + qg) * D_ + hh * HD_;
#pragma unroll
  for (int dt = 0; dt < 8; ++dt)
#pragma unroll
    for (int r = 0; r < 8; ++r)
      orow[dt * 16 + 8 * h + r] = f2bf(o[dt][r] * inv);
}

// ------------------------------------------------------- output projection
// Same blocking as QKV GEMM: LDS-staged A, 32x32 per wave, f32 store.
__global__ __launch_bounds__(256)
void wo_gemm_kernel(const unsigned short* __restrict__ ab,   // [B*S, D] bf16
                    const unsigned short* __restrict__ wob,  // [D, D] bf16
                    float* __restrict__ out) {               // [B*S, D] f32
  __shared__ __align__(16) unsigned short ldsA[2][64][128];

  const int tid  = threadIdx.x;
  const int lane = tid & 31;
  const int wid  = tid >> 5;
  const int l16  = lane & 15;
  const int h    = lane >> 4;
  const int wm   = wid >> 2;
  const int wn   = wid & 3;

  const int m_blk = blockIdx.x * 64;
  const int n0w   = blockIdx.y * 128 + wn * 32;

  const unsigned short* brow0 = wob + (size_t)(n0w + l16) * D_;
  const unsigned short* brow1 = wob + (size_t)(n0w + 16 + l16) * D_;

  auto stage = [&](int buf, int k0) {
#pragma unroll
    for (int c = 0; c < 4; ++c) {
      const int chunk = tid * 4 + c;
      const int row   = chunk >> 3;
      const int off   = (chunk & 7) * 8;
      const unsigned short* g = ab + (size_t)(m_blk + row) * D_ + k0 + off;
      const unsigned laddr = (unsigned)(size_t)&ldsA[buf][row][off];
      async_copy16(laddr, g);
    }
  };

  v8f acc[2][2] = {};

  stage(0, 0);
  async_wait_all();
  __syncthreads();

  const int NS = D_ / 128;
  for (int s = 0; s < NS; ++s) {
    const int k0  = s * 128;
    const int cur = s & 1;
    if (s + 1 < NS) stage(cur ^ 1, k0 + 128);

#pragma unroll
    for (int kk = 0; kk < 128; kk += 32) {
      Frag16 a0, a1, b0, b1;
      const unsigned short* l0 = &ldsA[cur][wm * 32 + l16][kk];
      const unsigned short* l1 = &ldsA[cur][wm * 32 + 16 + l16][kk];
      a0.q[0] = *reinterpret_cast<const u32x4*>(l0 + 16 * h);
      a0.q[1] = *reinterpret_cast<const u32x4*>(l0 + 16 * h + 8);
      a1.q[0] = *reinterpret_cast<const u32x4*>(l1 + 16 * h);
      a1.q[1] = *reinterpret_cast<const u32x4*>(l1 + 16 * h + 8);
      b0.q[0] = *reinterpret_cast<const u32x4*>(brow0 + k0 + kk + 16 * h);
      b0.q[1] = *reinterpret_cast<const u32x4*>(brow0 + k0 + kk + 16 * h + 8);
      b1.q[0] = *reinterpret_cast<const u32x4*>(brow1 + k0 + kk + 16 * h);
      b1.q[1] = *reinterpret_cast<const u32x4*>(brow1 + k0 + kk + 16 * h + 8);
      acc[0][0] = WMMA_BF16(a0.v, b0.v, acc[0][0]);
      acc[0][1] = WMMA_BF16(a0.v, b1.v, acc[0][1]);
      acc[1][0] = WMMA_BF16(a1.v, b0.v, acc[1][0]);
      acc[1][1] = WMMA_BF16(a1.v, b1.v, acc[1][1]);
    }
    async_wait_all();
    __syncthreads();
  }

#pragma unroll
  for (int im = 0; im < 2; ++im)
#pragma unroll
    for (int in_ = 0; in_ < 2; ++in_)
#pragma unroll
      for (int r = 0; r < 8; ++r)
        out[(size_t)(m_blk + wm * 32 + im * 16 + r + 8 * h) * D_ +
            n0w + in_ * 16 + l16] = acc[im][in_][r];
}

// --------------------------------------------------------------- launcher
extern "C" void kernel_launch(void* const* d_in, const int* in_sizes, int n_in,
                              void* d_out, int out_size, void* d_ws, size_t ws_size,
                              hipStream_t stream) {
  const float* x  = (const float*)d_in[0];
  const float* fc = (const float*)d_in[1];
  const float* fs = (const float*)d_in[2];
  const float* wq = (const float*)d_in[3];
  const float* wk = (const float*)d_in[4];
  const float* wv = (const float*)d_in[5];
  const float* wo = (const float*)d_in[6];
  float* out = (float*)d_out;

  char* ws = (char*)d_ws;
  size_t off = 0;
  auto carve = [&](size_t bytes) -> char* {
    char* p = ws + off;
    off = (off + bytes + 255) & ~(size_t)255;
    return p;
  };
  unsigned short* xb   = (unsigned short*)carve((size_t)B_ * S_ * D_ * 2);
  unsigned short* wqb  = (unsigned short*)carve((size_t)D_ * D_ * 2);
  unsigned short* wkb  = (unsigned short*)carve((size_t)512 * D_ * 2);
  unsigned short* wvb  = (unsigned short*)carve((size_t)512 * D_ * 2);
  unsigned short* wob  = (unsigned short*)carve((size_t)D_ * D_ * 2);
  unsigned short* qbuf = (unsigned short*)carve((size_t)B_ * H_ * S_ * HD_ * 2);
  unsigned short* kbuf = (unsigned short*)carve((size_t)B_ * KV_ * S_ * HD_ * 2);
  unsigned short* vtb  = (unsigned short*)carve((size_t)B_ * KV_ * HD_ * S_ * 2);
  unsigned short* attb = (unsigned short*)carve((size_t)B_ * S_ * D_ * 2);
  (void)ws_size; (void)n_in; (void)in_sizes; (void)out_size;

  const int NX = B_ * S_ * D_;
  const int NQ = D_ * D_;
  const int NK = 512 * D_;
  cvt_bf16_kernel<<<2048, 256, 0, stream>>>(x,  xb,  NX);
  cvt_bf16_kernel<<<2048, 256, 0, stream>>>(wq, wqb, NQ);
  cvt_bf16_kernel<<<1024, 256, 0, stream>>>(wk, wkb, NK);
  cvt_bf16_kernel<<<1024, 256, 0, stream>>>(wv, wvb, NK);
  cvt_bf16_kernel<<<2048, 256, 0, stream>>>(wo, wob, NQ);

  qkv_rope_kernel<<<dim3(B_ * S_ / 64, 3072 / 128), 256, 0, stream>>>(
      xb, wqb, wkb, wvb, fc, fs, qbuf, kbuf, vtb);

  attn_fa_kernel<<<dim3(S_ / 64, B_ * H_), 128, 0, stream>>>(
      qbuf, kbuf, vtb, attb);

  wo_gemm_kernel<<<dim3(B_ * S_ / 64, D_ / 128), 256, 0, stream>>>(
      attb, wob, out);
}
